// MpnnGGNN_9655086481675
// MI455X (gfx1250) — matrix operand order repaired and checked by reading, not compiled, not run on hardware
//
#include <hip/hip_runtime.h>
#include <cstdint>
#include <cstddef>

#define B_ 8
#define N_ 1024
#define DIN_ 64
#define H_ 128
#define MSG_ 128
#define L_ 5
#define TGT_ 12
#define NODES_ (B_ * N_) /* 8192 */

typedef __attribute__((ext_vector_type(16))) __bf16 v16bf;
typedef __attribute__((ext_vector_type(8)))  __bf16 v8bf;
typedef __attribute__((ext_vector_type(8)))  float  v8f;
typedef __attribute__((ext_vector_type(4)))  float  f32x4;
typedef __attribute__((ext_vector_type(4)))  int    i32x4;

union U16bf { v16bf v; v8bf h[2]; __bf16 e[16]; };
union U8f   { v8f  v; float e[8]; };

#if defined(__HIP_DEVICE_COMPILE__) && __has_builtin(__builtin_amdgcn_global_load_async_to_lds_b128)
#define KAGG_ASYNC 1
typedef __attribute__((address_space(1))) i32x4 as1_i32x4;
typedef __attribute__((address_space(3))) i32x4 as3_i32x4;
#else
#define KAGG_ASYNC 0
#endif

__device__ __forceinline__ float sigmoidf_(float x) { return 1.0f / (1.0f + __expf(-x)); }

// B-fragment tile: 16 contiguous bf16 per lane (two b128 loads)
__device__ __forceinline__ void load_btile(U16bf& t, const __bf16* p) {
    t.h[0] = *(const v8bf*)p;
    t.h[1] = *(const v8bf*)(p + 8);
}
// A-fragment from row-major: two 8-elem chunks 16 elements apart
__device__ __forceinline__ void load_afrag(U16bf& t, const __bf16* p) {
    t.h[0] = *(const v8bf*)p;
    t.h[1] = *(const v8bf*)(p + 16);
}

// ---------------------------------------------------------------------------
__global__ void k_zero(float* __restrict__ out) {
    int i = threadIdx.x;
    if (i < B_ * TGT_) out[i] = 0.0f;
}

// ---------------------------------------------------------------------------
// Pre-swizzle a row-major f32 [K,N] weight matrix into bf16 B-fragment tiles.
// flat[(kt*ntiles+nt)*512 + lane*16 + e] : lane = n%16 + 16*h, klocal = e<8 ? 8h+e : 16+8h+(e-8)
__global__ void k_prep_w(const float* __restrict__ src, __bf16* __restrict__ dst,
                         int K, int N) {
    int idx = blockIdx.x * blockDim.x + threadIdx.x;
    int tot = K * N;
    if (idx >= tot) return;
    int e    = idx & 15;
    int lane = (idx >> 4) & 31;
    int t    = idx >> 9;
    int ntiles = N >> 4;
    int nt = t % ntiles;
    int kt = t / ntiles;
    int n = nt * 16 + (lane & 15);
    int k = kt * 32 + ((lane & 16) ? 8 : 0) + ((e < 8) ? e : 8 + e);
    dst[idx] = (__bf16)src[k * N + n];
}

// ---------------------------------------------------------------------------
__global__ void k_init(const float* __restrict__ h_in, float* __restrict__ h0f,
                       __bf16* __restrict__ h0b, float* __restrict__ nmask,
                       float* __restrict__ rmask) {
    int node = blockIdx.x;
    int t = threadIdx.x;
    float v = (t < DIN_) ? h_in[node * DIN_ + t] : 0.0f;
    __shared__ float sa[H_];
    __shared__ float ss[H_];
    sa[t] = fabsf(v); ss[t] = v;
    __syncthreads();
    for (int o = H_ / 2; o > 0; o >>= 1) {
        if (t < o) { sa[t] += sa[t + o]; ss[t] += ss[t + o]; }
        __syncthreads();
    }
    if (t == 0) {
        nmask[node] = sa[0] > 0.0f ? 1.0f : 0.0f;
        rmask[node] = ss[0] > 0.0f ? 1.0f : 0.0f;
    }
    h0f[node * H_ + t] = v;
    h0b[node * H_ + t] = (__bf16)v;
}

// ---------------------------------------------------------------------------
// Ml[l][b] = h[b] @ A[l]; output in pre-swizzled B-fragment layout.
// Double-buffered A/B fragments so loads for kt+1 overlap WMMAs for kt.
__global__ __launch_bounds__(256) void k_ml(const __bf16* __restrict__ hbf,
                                            const __bf16* __restrict__ wAF,
                                            __bf16* __restrict__ ml) {
    int gtid = blockIdx.x * blockDim.x + threadIdx.x;
    int wid  = gtid >> 5;
    int lane = threadIdx.x & 31;
    if (wid >= 512 * L_) return;
    int l  = wid % L_;
    int rt = wid / L_;

    int row   = rt * 16 + (lane & 15);
    int cbase = (lane & 16) ? 8 : 0;
    const __bf16* abase = hbf + (size_t)row * H_ + cbase;
    const __bf16* wbase = wAF + (size_t)l * H_ * MSG_ + lane * 16;

    U16bf a[2];
    U16bf b[2][8];
    load_afrag(a[0], abase);
    #pragma unroll
    for (int nt = 0; nt < 8; ++nt) load_btile(b[0][nt], wbase + (size_t)nt * 512);

    U8f acc[8];
    #pragma unroll
    for (int i = 0; i < 8; ++i)
        #pragma unroll
        for (int r = 0; r < 8; ++r) acc[i].e[r] = 0.0f;

    #pragma unroll
    for (int kt = 0; kt < 4; ++kt) {
        int cur = kt & 1, nxt = cur ^ 1;
        if (kt < 3) {
            load_afrag(a[nxt], abase + (kt + 1) * 32);
            #pragma unroll
            for (int nt = 0; nt < 8; ++nt)
                load_btile(b[nxt][nt], wbase + (size_t)((kt + 1) * 8 + nt) * 512);
        }
        #pragma unroll
        for (int nt = 0; nt < 8; ++nt)
            acc[nt].v = __builtin_amdgcn_wmma_f32_16x16x32_bf16(
                false, a[cur].v, false, b[cur][nt].v, (short)0, acc[nt].v, false, false);
    }

    // Epilogue: per thread the 8 accumulator rows are contiguous in the
    // B-fragment layout -> one b128 store per nt.
    int b_   = rt >> 6;
    int wt   = rt & 63;
    int kt32 = wt >> 1;
    int nloc = lane & 15;
    int mofs = (wt & 1) ? 8 : 0;
    int dl   = nloc + ((lane & 16) ? 16 : 0);
    #pragma unroll
    for (int nt = 0; nt < 8; ++nt) {
        v8bf o;
        #pragma unroll
        for (int r = 0; r < 8; ++r) o[r] = (__bf16)acc[nt].e[r];
        *(v8bf*)(ml + (((size_t)(l * B_ + b_) * 32 + kt32) * 8 + nt) * 512 + dl * 16 + mofs) = o;
    }
}

// ---------------------------------------------------------------------------
// agg[b,v,:] = sum_l (g[b]*(e[b]==l)) @ Ml[l][b]
// 8 waves per block share the 40KB Ml k-slab via double-buffered LDS staged
// with global_load_async_to_lds_b128 (ASYNCcnt), one barrier per k-step.
// wave w: vt = w&3 (16-row v-tile within 64-row block), nhalf = w>>2.
__global__ __launch_bounds__(256) void k_agg(const float* __restrict__ g,
                                             const int* __restrict__ e,
                                             const __bf16* __restrict__ ml,
                                             __bf16* __restrict__ aggbf) {
    extern __shared__ __align__(16) char smem[];
    const int tid   = threadIdx.x;
    const int lane  = tid & 31;
    const int w     = tid >> 5;
    const int b_    = blockIdx.x >> 4;
    const int vblk  = blockIdx.x & 15;
    const int vt    = w & 3;
    const int nhalf = w >> 2;
    const int v0    = vblk * 64 + vt * 16;
    const int vrow  = v0 + (lane & 15);
    const int cbase = (lane & 16) ? 8 : 0;
    const float* grow = g + ((size_t)(b_ * N_ + vrow)) * N_;
    const int*   erow = e + ((size_t)(b_ * N_ + vrow)) * N_;
    const char*  mlc  = (const char*)ml;

#if KAGG_ASYNC
    // Prologue: stage slab kt=0 into LDS buffer 0 (10 x 16B chunks/thread).
    #pragma unroll
    for (int c = 0; c < 10; ++c) {
        int ci = tid + 256 * c;         // 2560 chunks of 16B = 40KB
        int l = ci >> 9;
        int within = ci & 511;
        const char* src = mlc + ((size_t)((l * B_ + b_) * 32 + 0)) * 8192 + within * 16;
        char* dst = smem + l * 8192 + within * 16;
        __builtin_amdgcn_global_load_async_to_lds_b128((as1_i32x4*)src, (as3_i32x4*)dst, 0, 0);
    }
#endif

    // Preload g/e for kt=0
    f32x4 gc[4]; i32x4 ec[4];
    {
        int c = cbase;
        gc[0] = *(const f32x4*)(grow + c);      gc[1] = *(const f32x4*)(grow + c + 4);
        gc[2] = *(const f32x4*)(grow + c + 16); gc[3] = *(const f32x4*)(grow + c + 20);
        ec[0] = *(const i32x4*)(erow + c);      ec[1] = *(const i32x4*)(erow + c + 4);
        ec[2] = *(const i32x4*)(erow + c + 16); ec[3] = *(const i32x4*)(erow + c + 20);
    }

    U8f acc[4];
    #pragma unroll
    for (int j = 0; j < 4; ++j)
        #pragma unroll
        for (int r = 0; r < 8; ++r) acc[j].e[r] = 0.0f;

    #pragma unroll 2
    for (int kt = 0; kt < 32; ++kt) {
        // Prefetch next g/e into registers (clamped re-read on last iter).
        int ktn = (kt < 31) ? kt + 1 : 31;
        f32x4 gn[4]; i32x4 en[4];
        {
            int c = ktn * 32 + cbase;
            gn[0] = *(const f32x4*)(grow + c);      gn[1] = *(const f32x4*)(grow + c + 4);
            gn[2] = *(const f32x4*)(grow + c + 16); gn[3] = *(const f32x4*)(grow + c + 20);
            en[0] = *(const i32x4*)(erow + c);      en[1] = *(const i32x4*)(erow + c + 4);
            en[2] = *(const i32x4*)(erow + c + 16); en[3] = *(const i32x4*)(erow + c + 20);
        }

#if KAGG_ASYNC
        // Current slab ready: own async done + everyone arrived.
        asm volatile("s_wait_asynccnt 0x0" ::: "memory");
        __syncthreads();
        if (kt < 31) {   // stage slab kt+1 into the other buffer
            int bufn = (kt + 1) & 1;
            #pragma unroll
            for (int c = 0; c < 10; ++c) {
                int ci = tid + 256 * c;
                int l = ci >> 9;
                int within = ci & 511;
                const char* src =
                    mlc + ((size_t)((l * B_ + b_) * 32 + kt + 1)) * 8192 + within * 16;
                char* dst = smem + bufn * 40960 + l * 8192 + within * 16;
                __builtin_amdgcn_global_load_async_to_lds_b128((as1_i32x4*)src, (as3_i32x4*)dst, 0, 0);
            }
        }
#endif

        // Build the 5 label-masked bf16 A tiles from g/e.
        float gv[16]; int ev[16];
        #pragma unroll
        for (int i = 0; i < 4; ++i) {
            gv[i] = gc[0][i]; gv[4 + i] = gc[1][i]; gv[8 + i] = gc[2][i]; gv[12 + i] = gc[3][i];
            ev[i] = ec[0][i]; ev[4 + i] = ec[1][i]; ev[8 + i] = ec[2][i]; ev[12 + i] = ec[3][i];
        }
        U16bf aF[L_];
        #pragma unroll
        for (int l = 0; l < L_; ++l)
            #pragma unroll
            for (int i = 0; i < 16; ++i)
                aF[l].e[i] = (ev[i] == l) ? (__bf16)gv[i] : (__bf16)0.0f;

#if KAGG_ASYNC
        const char* lbase = smem + (kt & 1) * 40960;
        #pragma unroll
        for (int j = 0; j < 4; ++j) {
            int nt = nhalf * 4 + j;
            #pragma unroll
            for (int l = 0; l < L_; ++l) {
                const char* p = lbase + (size_t)(l * 8 + nt) * 1024 + lane * 32;
                U16bf bb;
                bb.h[0] = *(const v8bf*)p;
                bb.h[1] = *(const v8bf*)(p + 16);
                acc[j].v = __builtin_amdgcn_wmma_f32_16x16x32_bf16(
                    false, aF[l].v, false, bb.v, (short)0, acc[j].v, false, false);
            }
        }
#else
        #pragma unroll
        for (int j = 0; j < 4; ++j) {
            int nt = nhalf * 4 + j;
            #pragma unroll
            for (int l = 0; l < L_; ++l) {
                const __bf16* bp =
                    ml + (((size_t)(l * B_ + b_) * 32 + kt) * 8 + nt) * 512 + lane * 16;
                U16bf bb;
                load_btile(bb, bp);
                acc[j].v = __builtin_amdgcn_wmma_f32_16x16x32_bf16(
                    false, aF[l].v, false, bb.v, (short)0, acc[j].v, false, false);
            }
        }
#endif
        #pragma unroll
        for (int i = 0; i < 4; ++i) { gc[i] = gn[i]; ec[i] = en[i]; }
    }

    int nloc = lane & 15;
    int mofs = (lane & 16) ? 8 : 0;
    #pragma unroll
    for (int j = 0; j < 4; ++j)
        #pragma unroll
        for (int r = 0; r < 8; ++r) {
            int m = r + mofs;
            size_t row = (size_t)(b_ * N_ + v0 + m);
            int col = nhalf * 64 + j * 16 + nloc;
            aggbf[row * MSG_ + col] = (__bf16)acc[j].e[r];
        }
}

// ---------------------------------------------------------------------------
// GRU with double-buffered fragments: gi = agg@Wi + bi, gh = h@Wh + bh.
__global__ __launch_bounds__(256) void k_gru(
    const __bf16* __restrict__ aggbf, const __bf16* __restrict__ hbf,
    const float* __restrict__ hf, const __bf16* __restrict__ wiF,
    const __bf16* __restrict__ whF, const float* __restrict__ bi,
    const float* __restrict__ bh, const float* __restrict__ nmask,
    float* __restrict__ hf_next, __bf16* __restrict__ hbf_next) {
    int gtid = blockIdx.x * blockDim.x + threadIdx.x;
    int wid  = gtid >> 5;
    int lane = threadIdx.x & 31;
    if (wid >= 4096) return;
    int rt = wid >> 3;
    int nt = wid & 7;

    int row   = rt * 16 + (lane & 15);
    int cbase = (lane & 16) ? 8 : 0;
    const __bf16* pAa = aggbf + (size_t)row * MSG_ + cbase;
    const __bf16* pAh = hbf + (size_t)row * H_ + cbase;
    const __bf16* pWi = wiF + (size_t)lane * 16;
    const __bf16* pWh = whF + (size_t)lane * 16;

    U16bf aA[2], aH[2], bI[2][3], bH[2][3];
    load_afrag(aA[0], pAa);
    load_afrag(aH[0], pAh);
    #pragma unroll
    for (int s = 0; s < 3; ++s) {
        int tIdx = nt + 8 * s;
        load_btile(bI[0][s], pWi + (size_t)tIdx * 512);
        load_btile(bH[0][s], pWh + (size_t)tIdx * 512);
    }

    U8f accI[3], accH[3];
    #pragma unroll
    for (int s = 0; s < 3; ++s)
        #pragma unroll
        for (int r = 0; r < 8; ++r) { accI[s].e[r] = 0.0f; accH[s].e[r] = 0.0f; }

    #pragma unroll
    for (int kt = 0; kt < 4; ++kt) {
        int cur = kt & 1, nxt = cur ^ 1;
        if (kt < 3) {
            load_afrag(aA[nxt], pAa + (kt + 1) * 32);
            load_afrag(aH[nxt], pAh + (kt + 1) * 32);
            #pragma unroll
            for (int s = 0; s < 3; ++s) {
                int tIdx = (kt + 1) * 24 + nt + 8 * s;
                load_btile(bI[nxt][s], pWi + (size_t)tIdx * 512);
                load_btile(bH[nxt][s], pWh + (size_t)tIdx * 512);
            }
        }
        #pragma unroll
        for (int s = 0; s < 3; ++s) {
            accI[s].v = __builtin_amdgcn_wmma_f32_16x16x32_bf16(
                false, aA[cur].v, false, bI[cur][s].v, (short)0, accI[s].v, false, false);
            accH[s].v = __builtin_amdgcn_wmma_f32_16x16x32_bf16(
                false, aH[cur].v, false, bH[cur][s].v, (short)0, accH[s].v, false, false);
        }
    }

    int nloc = lane & 15;
    int mofs = (lane & 16) ? 8 : 0;
    #pragma unroll
    for (int r = 0; r < 8; ++r) {
        int m = r + mofs;
        size_t grow_ = (size_t)rt * 16 + m;
        int hcol = nt * 16 + nloc;
        float ir  = accI[0].e[r] + bi[hcol];
        float iz  = accI[1].e[r] + bi[H_ + hcol];
        float inn = accI[2].e[r] + bi[2 * H_ + hcol];
        float hr  = accH[0].e[r] + bh[hcol];
        float hz  = accH[1].e[r] + bh[H_ + hcol];
        float hn  = accH[2].e[r] + bh[2 * H_ + hcol];
        float rg = sigmoidf_(ir + hr);
        float z  = sigmoidf_(iz + hz);
        float nn = tanhf(inn + rg * hn);
        float hold = hf[grow_ * H_ + hcol];
        float hv = ((1.0f - z) * nn + z * hold) * nmask[grow_];
        hf_next[grow_ * H_ + hcol]  = hv;
        hbf_next[grow_ * H_ + hcol] = (__bf16)hv;
    }
}

// ---------------------------------------------------------------------------
__global__ __launch_bounds__(256) void k_readout(
    const float* __restrict__ hT, const float* __restrict__ h0,
    const float* __restrict__ Wg, const float* __restrict__ bg,
    const float* __restrict__ Wo, const float* __restrict__ bo,
    const float* __restrict__ rmask, float* __restrict__ out) {
    int gtid = blockIdx.x * blockDim.x + threadIdx.x;
    int wid  = gtid >> 5;
    int lane = threadIdx.x & 31;
    if (wid >= NODES_) return;
    int node = wid;
    int b_ = node >> 10;
    f32x4 ht4 = *(const f32x4*)(hT + (size_t)node * H_ + lane * 4);
    f32x4 h04 = *(const f32x4*)(h0 + (size_t)node * H_ + lane * 4);
    float rm = rmask[node];
    for (int t = 0; t < TGT_; ++t) {
        float gd = 0.0f, od = 0.0f;
        #pragma unroll
        for (int j = 0; j < 4; ++j) {
            int k = lane * 4 + j;
            gd += ht4[j] * Wg[k * TGT_ + t] + h04[j] * Wg[(H_ + k) * TGT_ + t];
            od += ht4[j] * Wo[k * TGT_ + t];
        }
        #pragma unroll
        for (int off = 16; off > 0; off >>= 1) {
            gd += __shfl_xor(gd, off, 32);
            od += __shfl_xor(od, off, 32);
        }
        if (lane == 0) {
            float gate = sigmoidf_(gd + bg[t]);
            atomicAdd(out + b_ * TGT_ + t, gate * (od + bo[t]) * rm);
        }
    }
}

// ---------------------------------------------------------------------------
extern "C" void kernel_launch(void* const* d_in, const int* in_sizes, int n_in,
                              void* d_out, int out_size, void* d_ws, size_t ws_size,
                              hipStream_t stream) {
    const float* g    = (const float*)d_in[0];
    const float* h_in = (const float*)d_in[1];
    const int*   e    = (const int*)d_in[2];
    const float* A    = (const float*)d_in[3];
    const float* Wi   = (const float*)d_in[4];
    const float* Wh   = (const float*)d_in[5];
    const float* bi   = (const float*)d_in[6];
    const float* bh   = (const float*)d_in[7];
    const float* Wg   = (const float*)d_in[8];
    const float* bg   = (const float*)d_in[9];
    const float* Wo   = (const float*)d_in[10];
    const float* bo   = (const float*)d_in[11];
    float* out = (float*)d_out;

    char* ws = (char*)d_ws;
    size_t o = 0;
    auto take = [&](size_t bytes) {
        size_t r = o;
        o += (bytes + 255) & ~(size_t)255;
        return r;
    };
    float*  h0f  = (float*)(ws + take((size_t)NODES_ * H_ * 4));
    float*  h1f  = (float*)(ws + take((size_t)NODES_ * H_ * 4));
    float*  h2f  = (float*)(ws + take((size_t)NODES_ * H_ * 4));
    __bf16* h0b  = (__bf16*)(ws + take((size_t)NODES_ * H_ * 2));
    __bf16* h1b  = (__bf16*)(ws + take((size_t)NODES_ * H_ * 2));
    __bf16* h2b  = (__bf16*)(ws + take((size_t)NODES_ * H_ * 2));
    __bf16* aggb = (__bf16*)(ws + take((size_t)NODES_ * MSG_ * 2));
    __bf16* ml   = (__bf16*)(ws + take((size_t)L_ * B_ * 32 * 8 * 512 * 2));
    __bf16* wAF  = (__bf16*)(ws + take((size_t)L_ * H_ * MSG_ * 2));
    __bf16* wiF  = (__bf16*)(ws + take((size_t)MSG_ * 3 * H_ * 2));
    __bf16* whF  = (__bf16*)(ws + take((size_t)H_ * 3 * H_ * 2));
    float*  nm   = (float*)(ws + take((size_t)NODES_ * 4));
    float*  rm   = (float*)(ws + take((size_t)NODES_ * 4));

    k_zero<<<1, 128, 0, stream>>>(out);
    k_init<<<NODES_, H_, 0, stream>>>(h_in, h0f, h0b, nm, rm);
    for (int l = 0; l < L_; ++l)
        k_prep_w<<<64, 256, 0, stream>>>(A + (size_t)l * H_ * MSG_,
                                         wAF + (size_t)l * H_ * MSG_, H_, MSG_);
    k_prep_w<<<192, 256, 0, stream>>>(Wi, wiF, MSG_, 3 * H_);
    k_prep_w<<<192, 256, 0, stream>>>(Wh, whF, H_, 3 * H_);

    const __bf16* curb = h0b;
    const float*  curf = h0f;
    __bf16* nb = h1b;  float* nf = h1f;
    __bf16* ob = h2b;  float* of = h2f;
    for (int layer = 0; layer < 3; ++layer) {
        k_ml<<<320, 256, 0, stream>>>(curb, wAF, ml);              // 2560 waves
        k_agg<<<128, 256, 81920, stream>>>(g, e, ml, aggb);        // 8 waves/block + LDS slab
        k_gru<<<512, 256, 0, stream>>>(aggb, curb, curf, wiF, whF, bi, bh, nm, nf, nb);
        curb = nb; curf = nf;
        __bf16* tb = nb; nb = ob; ob = tb;
        float* tf = nf; nf = of; of = tf;
    }

    k_readout<<<1024, 256, 0, stream>>>(curf, h0f, Wg, bg, Wo, bo, rm, out);
}
